// SelfAttention_55284819034889
// MI455X (gfx1250) — matrix operand order repaired and checked by reading, not compiled, not run on hardware
//
#include <hip/hip_runtime.h>
#include <hip/hip_bf16.h>

// Problem constants (match reference)
#define B_  4
#define S_  4096
#define E_  1024
#define HD_ 64

typedef _Float16 half_t;
typedef __attribute__((ext_vector_type(16))) _Float16 v16h;
typedef __attribute__((ext_vector_type(8)))  _Float16 v8h;
typedef __attribute__((ext_vector_type(8)))  float    v8f;

__device__ __forceinline__ v8f wmma_f16(v16h a, v16h b, v8f c) {
  // D = A(16x32 f16) * B(32x16 f16) + C(16x16 f32)
  return __builtin_amdgcn_wmma_f32_16x16x32_f16(
      /*neg_a=*/false, a, /*neg_b=*/false, b,
      /*c_mod=*/(short)0, c, /*reuse_a=*/false, /*reuse_b=*/false);
}

// ---------------------------------------------------------------------------
// Kernel 0: W [E,HD] f32  ->  Wt [HD,E] f16  (transpose + downconvert)
// ---------------------------------------------------------------------------
__global__ __launch_bounds__(256) void w_transpose_kernel(
    const float* __restrict__ W, half_t* __restrict__ Wt) {
  int idx = blockIdx.x * 256 + threadIdx.x;   // over HD_*E_
  int n = idx / E_;                           // hd column
  int k = idx % E_;                           // embed row
  Wt[idx] = (_Float16)W[k * HD_ + n];
}

// ---------------------------------------------------------------------------
// Kernel 1: projection  Out = X[B,S,E] @ W[E,HD] + bias  (f16 out)
// grid = (S/(16*8), B), block = 256 (8 waves). Each wave owns one 16-row
// s-tile and computes ALL HD=64 columns (4 accumulators), so the fp32 X slab
// is loaded and converted exactly once. W (f16, 128KB) is L2-resident.
// TRANS==1 stores Out as [B,HD,S] (for V).
// ---------------------------------------------------------------------------
template <int TRANS>
__global__ __launch_bounds__(256) void qkv_proj_kernel(
    const float* __restrict__ X, const half_t* __restrict__ Wt,
    const float* __restrict__ bias, half_t* __restrict__ Out) {
  const int lane  = threadIdx.x & 31;
  const int wave  = threadIdx.x >> 5;
  const int col   = lane & 15;     // N index (and A row index)
  const int khalf = lane >> 4;     // which K half this lane holds
  const int s_base = (blockIdx.x * 8 + wave) * 16;
  const int b      = blockIdx.y;

  // A source: row (s_base+col) of X; K pattern per ISA: j<8 -> K=8*khalf+j,
  // j>=8 -> K=16+8*khalf+(j-8)
  const float* xrow = X + ((size_t)b * S_ + s_base + col) * E_;
  // B source: Wt[n, e] contiguous in e; lane holds same K pattern as A
  const half_t* wbase = Wt + (size_t)col * E_;

  v8f acc[4] = {{}, {}, {}, {}};
  for (int e0 = 0; e0 < E_; e0 += 32) {
    const float4 a0 = *(const float4*)(xrow + e0 + 8 * khalf);
    const float4 a1 = *(const float4*)(xrow + e0 + 8 * khalf + 4);
    const float4 a2 = *(const float4*)(xrow + e0 + 16 + 8 * khalf);
    const float4 a3 = *(const float4*)(xrow + e0 + 16 + 8 * khalf + 4);
    v16h A;
    A[0]=(_Float16)a0.x; A[1]=(_Float16)a0.y; A[2]=(_Float16)a0.z; A[3]=(_Float16)a0.w;
    A[4]=(_Float16)a1.x; A[5]=(_Float16)a1.y; A[6]=(_Float16)a1.z; A[7]=(_Float16)a1.w;
    A[8]=(_Float16)a2.x; A[9]=(_Float16)a2.y; A[10]=(_Float16)a2.z; A[11]=(_Float16)a2.w;
    A[12]=(_Float16)a3.x; A[13]=(_Float16)a3.y; A[14]=(_Float16)a3.z; A[15]=(_Float16)a3.w;

#pragma unroll
    for (int t = 0; t < 4; ++t) {
      const half_t* wrow = wbase + (size_t)(16 * t) * E_ + e0 + 8 * khalf;
      const v8h w0 = *(const v8h*)(wrow);
      const v8h w1 = *(const v8h*)(wrow + 16);
      v16h Bm;
#pragma unroll
      for (int i = 0; i < 8; ++i) { Bm[i] = w0[i]; Bm[i + 8] = w1[i]; }
      acc[t] = wmma_f16(A, Bm, acc[t]);
    }
  }

#pragma unroll
  for (int t = 0; t < 4; ++t) {
    const int hd = 16 * t + col;
    const float bval = bias[hd];
#pragma unroll
    for (int r = 0; r < 8; ++r) {
      const float v = acc[t][r] + bval;
      const int m = r + 8 * khalf;                    // row within tile
      if (TRANS == 0) {
        Out[((size_t)b * S_ + s_base + m) * HD_ + hd] = (_Float16)v;
      } else {                                        // [B, HD, S]
        Out[(size_t)b * HD_ * S_ + (size_t)hd * S_ + s_base + m] = (_Float16)v;
      }
    }
  }
}

// ---------------------------------------------------------------------------
// Kernel 2: flash attention.
// 8 waves/block, each wave owns one 16-row query tile; loops over S keys in
// chunks of 32. Online softmax, O accumulated in 4x v8f (16x64 f32).
// Mask (256MB, the traffic floor) is streamed once with NT loads so it does
// not evict the 4MB of f16 K/V living in the 192MB L2.
// ---------------------------------------------------------------------------
__global__ __launch_bounds__(256) void flash_attn_kernel(
    const half_t* __restrict__ Qh,   // [B,S,HD]
    const half_t* __restrict__ Kh,   // [B,S,HD]
    const half_t* __restrict__ Vt,   // [B,HD,S]
    const float*  __restrict__ mask, // [B,S,S]
    float* __restrict__ out) {       // [B,S,HD]
  const int lane  = threadIdx.x & 31;
  const int wave  = threadIdx.x >> 5;
  const int col   = lane & 15;
  const int khalf = lane >> 4;

  const int wg     = blockIdx.x * 8 + wave;   // global query-tile id
  const int b      = wg / (S_ / 16);
  const int q_base = (wg % (S_ / 16)) * 16;

  __shared__ __align__(16) _Float16 pbuf[8][16 * 32];  // per-wave P staging
  _Float16* pb = pbuf[wave];

  // Preload Q as two A-operands (HD split K=0..31, K=32..63), resident.
  const half_t* qrow = Qh + ((size_t)b * S_ + q_base + col) * HD_;
  const v8h q00 = *(const v8h*)(qrow + 8 * khalf);
  const v8h q01 = *(const v8h*)(qrow + 16 + 8 * khalf);
  const v8h q10 = *(const v8h*)(qrow + 32 + 8 * khalf);
  const v8h q11 = *(const v8h*)(qrow + 48 + 8 * khalf);
  v16h aQ0, aQ1;
#pragma unroll
  for (int i = 0; i < 8; ++i) {
    aQ0[i] = q00[i]; aQ0[i + 8] = q01[i];
    aQ1[i] = q10[i]; aQ1[i + 8] = q11[i];
  }

  v8f   Oacc[4] = {{}, {}, {}, {}};
  float mrow[8], lrow[8];
#pragma unroll
  for (int r = 0; r < 8; ++r) { mrow[r] = -INFINITY; lrow[r] = 0.0f; }

  const float*  maskB = mask + (size_t)b * S_ * S_;
  const half_t* Kb    = Kh + (size_t)b * S_ * HD_;
  const half_t* Vb    = Vt + (size_t)b * HD_ * S_;

  for (int kb = 0; kb < S_; kb += 32) {
    // ---- logits: two 16x16 tiles (keys kb..+15, kb+16..+31), K-dim = HD=64
    v8f c[2];
#pragma unroll
    for (int kt = 0; kt < 2; ++kt) {
      const int key = kb + kt * 16 + col;     // B-operand column
      const half_t* krow = Kb + (size_t)key * HD_;
      const v8h k0a = *(const v8h*)(krow + 8 * khalf);
      const v8h k0b = *(const v8h*)(krow + 16 + 8 * khalf);
      const v8h k1a = *(const v8h*)(krow + 32 + 8 * khalf);
      const v8h k1b = *(const v8h*)(krow + 48 + 8 * khalf);
      v16h bK0, bK1;
#pragma unroll
      for (int i = 0; i < 8; ++i) {
        bK0[i] = k0a[i]; bK0[i + 8] = k0b[i];
        bK1[i] = k1a[i]; bK1[i + 8] = k1b[i];
      }
      v8f cc = {};
      cc = wmma_f16(aQ0, bK0, cc);
      cc = wmma_f16(aQ1, bK1, cc);
      c[kt] = cc;
    }

    // ---- scale + additive mask + online softmax (per-row, cross-lane)
    float fr[8];
#pragma unroll
    for (int r = 0; r < 8; ++r) {
      const int rowg = q_base + r + 8 * khalf;
      const float* mrowp = maskB + (size_t)rowg * S_ + kb;
      if (r == 0) __builtin_prefetch(mrowp + S_ * 8 + 32, 0, 0);
      const float m0 = __builtin_nontemporal_load(mrowp + col);
      const float m1 = __builtin_nontemporal_load(mrowp + 16 + col);
      float x0 = c[0][r] * 0.125f - 1.0e9f * m0;
      float x1 = c[1][r] * 0.125f - 1.0e9f * m1;

      float t = fmaxf(x0, x1);
      t = fmaxf(t, __shfl_xor(t, 1, 32));
      t = fmaxf(t, __shfl_xor(t, 2, 32));
      t = fmaxf(t, __shfl_xor(t, 4, 32));
      t = fmaxf(t, __shfl_xor(t, 8, 32));   // max over 16 key lanes

      const float mn = fmaxf(mrow[r], t);
      const float f  = __expf(mrow[r] - mn);
      const float p0 = __expf(x0 - mn);
      const float p1 = __expf(x1 - mn);
      float s = p0 + p1;
      s += __shfl_xor(s, 1, 32);
      s += __shfl_xor(s, 2, 32);
      s += __shfl_xor(s, 4, 32);
      s += __shfl_xor(s, 8, 32);

      lrow[r] = lrow[r] * f + s;
      mrow[r] = mn;
      fr[r]   = f;

      // stash P (C-layout) into LDS for re-read in A-layout
      pb[(r + 8 * khalf) * 32 + col]      = (_Float16)p0;
      pb[(r + 8 * khalf) * 32 + 16 + col] = (_Float16)p1;
    }
#pragma unroll
    for (int t = 0; t < 4; ++t)
#pragma unroll
      for (int r = 0; r < 8; ++r) Oacc[t][r] *= fr[r];

    // ---- P (16x32, A-layout) from LDS
    const v8h plo = *(const v8h*)(pb + col * 32 + 8 * khalf);
    const v8h phi = *(const v8h*)(pb + col * 32 + 16 + 8 * khalf);
    v16h pA;
#pragma unroll
    for (int i = 0; i < 8; ++i) { pA[i] = plo[i]; pA[i + 8] = phi[i]; }

    // ---- O += P @ V  (4 hd tiles of 16, K-dim = 32 keys)
#pragma unroll
    for (int t = 0; t < 4; ++t) {
      const half_t* vrow = Vb + (size_t)(16 * t + col) * S_ + kb + 8 * khalf;
      const v8h v0 = *(const v8h*)vrow;
      const v8h v1 = *(const v8h*)(vrow + 16);
      v16h bV;
#pragma unroll
      for (int i = 0; i < 8; ++i) { bV[i] = v0[i]; bV[i + 8] = v1[i]; }
      Oacc[t] = wmma_f16(pA, bV, Oacc[t]);
    }
  }

  // ---- normalize and store
  float inv[8];
#pragma unroll
  for (int r = 0; r < 8; ++r) inv[r] = 1.0f / lrow[r];
#pragma unroll
  for (int t = 0; t < 4; ++t)
#pragma unroll
    for (int r = 0; r < 8; ++r) {
      float* dst =
          out + ((size_t)b * S_ + q_base + r + 8 * khalf) * HD_ + 16 * t + col;
      __builtin_nontemporal_store(Oacc[t][r] * inv[r], dst);
    }
}

// ---------------------------------------------------------------------------
extern "C" void kernel_launch(void* const* d_in, const int* in_sizes, int n_in,
                              void* d_out, int out_size, void* d_ws,
                              size_t ws_size, hipStream_t stream) {
  (void)in_sizes; (void)n_in; (void)out_size; (void)ws_size;
  const float* values = (const float*)d_in[0];
  const float* keys   = (const float*)d_in[1];
  const float* query  = (const float*)d_in[2];
  const float* mask   = (const float*)d_in[3];
  const float* Wq = (const float*)d_in[4];
  const float* bq = (const float*)d_in[5];
  const float* Wk = (const float*)d_in[6];
  const float* bk = (const float*)d_in[7];
  const float* Wv = (const float*)d_in[8];
  const float* bv = (const float*)d_in[9];
  float* out = (float*)d_out;

  // Workspace carve-up (f16): 3x Wt[HD,E] + Q,K [B,S,HD] + Vt [B,HD,S] ≈ 6.7MB
  half_t* ws = (half_t*)d_ws;
  size_t o = 0;
  half_t* Wqt = ws + o; o += (size_t)HD_ * E_;
  half_t* Wkt = ws + o; o += (size_t)HD_ * E_;
  half_t* Wvt = ws + o; o += (size_t)HD_ * E_;
  half_t* Qh  = ws + o; o += (size_t)B_ * S_ * HD_;
  half_t* Kh  = ws + o; o += (size_t)B_ * S_ * HD_;
  half_t* Vth = ws + o; o += (size_t)B_ * S_ * HD_;

  w_transpose_kernel<<<(HD_ * E_) / 256, 256, 0, stream>>>(Wq, Wqt);
  w_transpose_kernel<<<(HD_ * E_) / 256, 256, 0, stream>>>(Wk, Wkt);
  w_transpose_kernel<<<(HD_ * E_) / 256, 256, 0, stream>>>(Wv, Wvt);

  dim3 pg(S_ / (16 * 8), B_);
  qkv_proj_kernel<0><<<pg, 256, 0, stream>>>(query,  Wqt, bq, Qh);
  qkv_proj_kernel<0><<<pg, 256, 0, stream>>>(keys,   Wkt, bk, Kh);
  qkv_proj_kernel<1><<<pg, 256, 0, stream>>>(values, Wvt, bv, Vth);

  flash_attn_kernel<<<(B_ * S_ / 16) / 8, 256, 0, stream>>>(Qh, Kh, Vth, mask,
                                                            out);
}
